// MLA_58626303591024
// MI455X (gfx1250) — matrix operand order repaired and checked by reading, not compiled, not run on hardware
//
#include <hip/hip_runtime.h>
#include <hip/hip_bf16.h>
#include <math.h>

// ---------------- problem constants (match reference) ----------------
#define BATCH   2
#define SEQ     2048
#define DIM     2048
#define NH      16
#define D_NOPE  128
#define D_ROPE  64
#define D_V     128
#define D_QK    192      // D_NOPE + D_ROPE
#define KV_RANK 512
#define D_EFF   576      // KV_RANK + D_ROPE

typedef __attribute__((ext_vector_type(16))) __bf16 v16bf;
typedef __attribute__((ext_vector_type(8)))  float  v8f;

#if defined(__gfx1250__) && __has_builtin(__builtin_amdgcn_tensor_load_to_lds)
#define USE_TDM 1
#else
#define USE_TDM 0
#endif

#if USE_TDM
typedef __attribute__((ext_vector_type(4))) unsigned int v4u;
typedef __attribute__((ext_vector_type(8))) int v8i;
typedef __attribute__((ext_vector_type(4))) int v4i;

// Issue a TDM 2-D tile load: tile_rows x tile_cols elements (2 bytes each) from
// a row-major global tensor (row stride = row_stride_elems) into packed LDS at
// lds_off. Rows >= tensor_rows / cols >= tensor_cols are zero-filled by the TDM.
__device__ __forceinline__ void tdm_load_2d(unsigned lds_off, const void* gptr,
                                            unsigned tile_cols, unsigned tensor_cols,
                                            unsigned tile_rows, unsigned tensor_rows,
                                            unsigned row_stride_elems) {
  const unsigned long long ga = (unsigned long long)(size_t)gptr;
  v4u g0;
  g0[0] = 1u;                                                    // count=1, user D#
  g0[1] = lds_off;                                               // lds_addr (bytes)
  g0[2] = (unsigned)ga;                                          // global_addr[31:0]
  g0[3] = (unsigned)((ga >> 32) & 0x01ffffffull) | (2u << 30);   // addr[56:32] | type=2
  v8i g1;
  g1[0] = (int)(1u << 16);                                       // data_size = 2 bytes
  g1[1] = (int)(tensor_cols << 16);                              // tensor_dim0[15:0] @bit48
  g1[2] = (int)((tensor_cols >> 16) | (tensor_rows << 16));      // dim0 hi | tensor_dim1 lo
  g1[3] = (int)((tensor_rows >> 16) | (tile_cols << 16));        // dim1 hi | tile_dim0
  g1[4] = (int)(tile_rows & 0xffffu);                            // tile_dim1 (tile_dim2=0)
  g1[5] = (int)row_stride_elems;                                 // tensor_dim0_stride lo32
  g1[6] = 0;
  g1[7] = 0;
  const v4i z4 = {0, 0, 0, 0};
  const v8i z8 = {0, 0, 0, 0, 0, 0, 0, 0};
  // amdgpu-toolchain (clang-23) 6-arg form: (g0, g1, g2, g3, extra, cpol)
  __builtin_amdgcn_tensor_load_to_lds(g0, g1, z4, z4, z8, 0);
}
#endif

__device__ __forceinline__ v8f vzero8() {
  v8f z;
#pragma unroll
  for (int i = 0; i < 8; ++i) z[i] = 0.0f;
  return z;
}

// A-fragment (16x32 bf16, M x K): lanes 0-15 row M, K in {0..7,16..23};
// lanes 16-31 same rows, K in {8..15,24..31}. element [row][k] = base[row*rowStride + k]
__device__ __forceinline__ v16bf load_frag_a(const __bf16* base, int rowStride) {
  const int lane = threadIdx.x & 31;
  const int half = lane >> 4;
  const int row  = lane & 15;
  v16bf f;
#pragma unroll
  for (int v = 0; v < 8; ++v) {
    const int k = ((v < 4) ? (2 * v) : (16 + 2 * (v - 4))) + half * 8;
    const __bf16* p = base + row * rowStride + k;
    f[2 * v]     = p[0];
    f[2 * v + 1] = p[1];
  }
  return f;
}

// B-fragment (32x16 bf16, K x N): N in lane%16, lanes 0-15 hold K {0..7,16..23},
// lanes 16-31 hold K {8..15,24..31}. element [k][n] = base[k*strideK + n*strideN]
__device__ __forceinline__ v16bf load_frag_b(const __bf16* base, int strideK, int strideN) {
  const int lane = threadIdx.x & 31;
  const int half = lane >> 4;
  const int col  = lane & 15;
  v16bf f;
#pragma unroll
  for (int v = 0; v < 8; ++v) {
    const int k = ((v < 4) ? (2 * v) : (16 + 2 * (v - 4))) + half * 8;
    f[2 * v]     = base[k * strideK + col * strideN];
    f[2 * v + 1] = base[(k + 1) * strideK + col * strideN];
  }
  return f;
}

__device__ __forceinline__ v8f wmma_bf16(v16bf a, v16bf b, v8f c) {
  return __builtin_amdgcn_wmma_f32_16x16x32_bf16(false, a, false, b, (short)0, c, false, false);
}

// ---------------- fp32 -> bf16 conversion ----------------
__global__ void k_cvt_bf16(const float* __restrict__ src, __bf16* __restrict__ dst, int n) {
  int i = blockIdx.x * blockDim.x + threadIdx.x;
  const int stride = gridDim.x * blockDim.x;
  for (; i < n; i += stride) dst[i] = (__bf16)src[i];
}

// ---------------- generic batched WMMA GEMM, TDM double-buffered ----------------
// C(M,N) = A(M,K) * op(B). B_IS_KN=false: B is (N,K) row-major (NT, B^T).
// B_IS_KN=true: B is (K,N) row-major (NN). fp32 accumulate.
template <bool B_IS_KN, bool OUT_BF16>
__global__ __launch_bounds__(256) void k_gemm(
    const __bf16* __restrict__ A, const __bf16* __restrict__ B, void* __restrict__ C,
    int M, int N, int K, int lda, int ldb, int ldc,
    long long strideA, long long strideB, long long strideC) {
  constexpr int BM = 128, BN = 128, BK = 32;
  __shared__ __bf16 As[2][BM * BK];
  __shared__ __bf16 Bs[2][BN * BK];  // NT: [n][k]; NN: [k][n]
  const int tid  = threadIdx.x;
  const int lane = tid & 31;
  const int wid  = tid >> 5;
  const int wm0  = (wid & 3) * 32;   // 4 waves along M
  const int wn0  = (wid >> 2) * 64;  // 2 waves along N
  const int m0   = blockIdx.y * BM;
  const int n0   = blockIdx.x * BN;
  A += (long long)blockIdx.z * strideA;
  B += (long long)blockIdx.z * strideB;
  char* Cb = (char*)C + (long long)blockIdx.z * strideC * (OUT_BF16 ? 2 : 4);

  v8f acc[2][4];
#pragma unroll
  for (int i = 0; i < 2; ++i)
#pragma unroll
    for (int j = 0; j < 4; ++j) acc[i][j] = vzero8();

#if USE_TDM
  // Issue both tiles of one K-step via the Tensor Data Mover (OOB rows/cols
  // beyond the tensor dims are zero-filled by the TDM).
  auto issue = [&](int k0, int buf) {
    if (tid == 0) {
      tdm_load_2d((unsigned)(size_t)&As[buf][0], A + (long long)m0 * lda + k0,
                  BK, BK, BM, (unsigned)(M - m0), (unsigned)lda);
      if (B_IS_KN)
        tdm_load_2d((unsigned)(size_t)&Bs[buf][0], B + (long long)k0 * ldb + n0,
                    BN, (unsigned)(N - n0), BK, BK, (unsigned)ldb);
      else
        tdm_load_2d((unsigned)(size_t)&Bs[buf][0], B + (long long)n0 * ldb + k0,
                    BK, BK, BN, (unsigned)(N - n0), (unsigned)ldb);
    }
  };
  int cur = 0;
  issue(0, 0);
#else
  auto stage = [&](int k0, int buf) {
    for (int it = 0; it < BM * BK / (4 * 256); ++it) {
      const int i = it * 256 + tid;
      const int idx = i * 4;
      const int r = idx >> 5, c = idx & 31;
      uint2 val = make_uint2(0u, 0u);
      const int gm = m0 + r;
      if (gm < M) val = *(const uint2*)(A + (long long)gm * lda + (k0 + c));
      *(uint2*)(&As[buf][r * BK + c]) = val;
    }
    if (B_IS_KN) {
      for (int it = 0; it < BK * BN / (4 * 256); ++it) {
        const int i = it * 256 + tid;
        const int idx = i * 4;
        const int r = idx >> 7, c = idx & 127;  // r = k, c = n
        uint2 val = make_uint2(0u, 0u);
        const int gn = n0 + c;
        if (gn < N) val = *(const uint2*)(B + (long long)(k0 + r) * ldb + gn);
        *(uint2*)(&Bs[buf][r * BN + c]) = val;
      }
    } else {
      for (int it = 0; it < BN * BK / (4 * 256); ++it) {
        const int i = it * 256 + tid;
        const int idx = i * 4;
        const int r = idx >> 5, c = idx & 31;  // r = n, c = k
        uint2 val = make_uint2(0u, 0u);
        const int gn = n0 + r;
        if (gn < N) val = *(const uint2*)(B + (long long)gn * ldb + (k0 + c));
        *(uint2*)(&Bs[buf][r * BK + c]) = val;
      }
    }
  };
  int cur = 0;
  stage(0, 0);
#endif

  for (int k0 = 0; k0 < K; k0 += BK) {
    const bool has_next = (k0 + BK) < K;
#if USE_TDM
    if (has_next) issue(k0 + BK, cur ^ 1);
    if (tid == 0) {
      // TDM ops complete in issue order: <=2 outstanding means this tile is done.
      if (has_next) __builtin_amdgcn_s_wait_tensorcnt(2);
      else          __builtin_amdgcn_s_wait_tensorcnt(0);
    }
    __syncthreads();
#else
    __syncthreads();
#endif

    v16bf af[2];
    af[0] = load_frag_a(&As[cur][wm0 * BK], BK);
    af[1] = load_frag_a(&As[cur][(wm0 + 16) * BK], BK);
    v16bf bfm[4];
#pragma unroll
    for (int in = 0; in < 4; ++in) {
      if (B_IS_KN)
        bfm[in] = load_frag_b(&Bs[cur][wn0 + in * 16], BN, 1);
      else
        bfm[in] = load_frag_b(&Bs[cur][(wn0 + in * 16) * BK], 1, BK);
    }
#pragma unroll
    for (int im = 0; im < 2; ++im)
#pragma unroll
      for (int in = 0; in < 4; ++in) acc[im][in] = wmma_bf16(af[im], bfm[in], acc[im][in]);
    __syncthreads();
#if !USE_TDM
    if (has_next) stage(k0 + BK, cur ^ 1);
#endif
    cur ^= 1;
  }

  const int half = lane >> 4, cn = lane & 15;
#pragma unroll
  for (int im = 0; im < 2; ++im)
#pragma unroll
    for (int in = 0; in < 4; ++in)
#pragma unroll
      for (int r = 0; r < 8; ++r) {
        const int gm = m0 + wm0 + im * 16 + r + 8 * half;
        const int gn = n0 + wn0 + in * 16 + cn;
        if (gm < M && gn < N) {
          const float v = acc[im][in][r];
          if (OUT_BF16)
            ((__bf16*)Cb)[(long long)gm * ldc + gn] = (__bf16)v;
          else
            ((float*)Cb)[(long long)gm * ldc + gn] = v;
        }
      }
}

// ---------------- q prep: copy q_nope to bf16, RoPE q_pe into q_eff[512:576] ----------------
__global__ __launch_bounds__(128) void k_prep_q(
    const float* __restrict__ qf, const float* __restrict__ cosb,
    const float* __restrict__ sinb, __bf16* __restrict__ qeff, __bf16* __restrict__ qn) {
  const int idx = blockIdx.x;  // (b*S + s)*H + h
  const int t = threadIdx.x;
  const float* row = qf + (long long)idx * D_QK;
  if (t < D_NOPE) qn[(long long)idx * D_NOPE + t] = (__bf16)row[t];
  if (t < D_ROPE / 2) {
    const int spos = (idx / NH) % SEQ;
    const float c  = cosb[spos * (D_ROPE / 2) + t];
    const float sn = sinb[spos * (D_ROPE / 2) + t];
    const float xr = row[D_NOPE + 2 * t];
    const float xi = row[D_NOPE + 2 * t + 1];
    __bf16* qe = qeff + (long long)idx * D_EFF + KV_RANK;
    qe[2 * t]     = (__bf16)(xr * c - xi * sn);
    qe[2 * t + 1] = (__bf16)(xr * sn + xi * c);
  }
}

// ---------------- k prep: RMSNorm(kv_c) and RoPE(k_pe) -> k_eff (576, bf16) ----------------
__global__ __launch_bounds__(128) void k_prep_k(
    const float* __restrict__ kvf, const float* __restrict__ cosb,
    const float* __restrict__ sinb, const float* __restrict__ wnorm,
    __bf16* __restrict__ keff) {
  const int m = blockIdx.x;  // b*S + t
  const int t = threadIdx.x;
  const float* row = kvf + (long long)m * D_EFF;
  float ss = 0.0f;
  for (int i = t; i < KV_RANK; i += 128) {
    const float v = row[i];
    ss += v * v;
  }
#pragma unroll
  for (int off = 16; off > 0; off >>= 1) ss += __shfl_xor(ss, off, 32);
  __shared__ float red[4];
  if ((t & 31) == 0) red[t >> 5] = ss;
  __syncthreads();
  const float tot  = red[0] + red[1] + red[2] + red[3];
  const float rinv = rsqrtf(tot / (float)KV_RANK + 1e-6f);
  __bf16* out = keff + (long long)m * D_EFF;
  for (int i = t; i < KV_RANK; i += 128) out[i] = (__bf16)(row[i] * rinv * wnorm[i]);
  if (t < D_ROPE / 2) {
    const int spos = m % SEQ;
    const float c  = cosb[spos * (D_ROPE / 2) + t];
    const float sn = sinb[spos * (D_ROPE / 2) + t];
    const float xr = row[KV_RANK + 2 * t];
    const float xi = row[KV_RANK + 2 * t + 1];
    out[KV_RANK + 2 * t]     = (__bf16)(xr * c - xi * sn);
    out[KV_RANK + 2 * t + 1] = (__bf16)(xr * sn + xi * c);
  }
}

// ---------------- flash attention over the 576-d latent space ----------------
// grid: (S/32, H, B); block 256 = 8 waves; wave (rg,cg): rg in {0,1} query row
// group (16 rows), cg in {0..3} value column group (128 of 512 cols).
__global__ __launch_bounds__(256) void k_attn(
    const __bf16* __restrict__ Qe, const __bf16* __restrict__ Ke,
    __bf16* __restrict__ O, float scale) {
  constexpr int LDK = D_EFF;  // packed rows (TDM writes tiles contiguously)
  __shared__ __bf16 Ks[32 * LDK];
  __shared__ __bf16 Ps[8][16 * 32];
  const int tid  = threadIdx.x;
  const int lane = tid & 31;
  const int wid  = tid >> 5;
  const int rg   = wid & 1;
  const int cg   = wid >> 1;
  const int half = lane >> 4;
  const int cn   = lane & 15;
  const int blk  = blockIdx.x;
  const int h    = blockIdx.y;
  const int b    = blockIdx.z;

  // stage this block's 32 Q rows through LDS, then pull A-fragments to registers
#if USE_TDM
  if (tid == 0) {
    tdm_load_2d((unsigned)(size_t)&Ks[0],
                Qe + (((long long)(b * SEQ + blk * 32)) * NH + h) * D_EFF,
                D_EFF, D_EFF, 32, 32, NH * D_EFF);
    __builtin_amdgcn_s_wait_tensorcnt(0);
  }
  __syncthreads();
#else
  for (int it = 0; it < 32 * D_EFF / (4 * 256); ++it) {
    const int i = it * 256 + tid;
    const int idx = i * 4;
    const int r = idx / D_EFF, c = idx % D_EFF;
    const __bf16* src = Qe + (((long long)(b * SEQ + blk * 32 + r)) * NH + h) * D_EFF + c;
    *(uint2*)(Ks + r * LDK + c) = *(const uint2*)src;
  }
  __syncthreads();
#endif
  v16bf qfr[D_EFF / 32];
#pragma unroll
  for (int kk = 0; kk < D_EFF / 32; ++kk)
    qfr[kk] = load_frag_a(Ks + (rg * 16) * LDK + kk * 32, LDK);
  __syncthreads();

  v8f o[8];
#pragma unroll
  for (int nt = 0; nt < 8; ++nt) o[nt] = vzero8();
  float mrow[8], lrow[8];
#pragma unroll
  for (int r = 0; r < 8; ++r) {
    mrow[r] = -3.0e38f;
    lrow[r] = 0.0f;
  }

  const int ntiles = blk + 1;  // causal: key tiles with tb <= blk*32
  for (int tile = 0; tile < ntiles; ++tile) {
    const int tb = tile * 32;
#if USE_TDM
    if (tid == 0) {
      tdm_load_2d((unsigned)(size_t)&Ks[0],
                  Ke + ((long long)(b * SEQ + tb)) * D_EFF,
                  D_EFF, D_EFF, 32, 32, D_EFF);
      __builtin_amdgcn_s_wait_tensorcnt(0);
    }
    __syncthreads();
#else
    for (int it = 0; it < 32 * D_EFF / (4 * 256); ++it) {
      const int i = it * 256 + tid;
      const int idx = i * 4;
      const int r = idx / D_EFF, c = idx % D_EFF;
      const __bf16* src = Ke + ((long long)(b * SEQ + tb + r)) * D_EFF + c;
      *(uint2*)(Ks + r * LDK + c) = *(const uint2*)src;
    }
    __syncthreads();
#endif

    // scores: S(16x32) = Q(16x576) . K_tile(32x576)^T
    v8f s0 = vzero8(), s1 = vzero8();
#pragma unroll
    for (int kk = 0; kk < D_EFF / 32; ++kk) {
      const v16bf b0 = load_frag_b(Ks + kk * 32, 1, LDK);            // t in 0..15
      const v16bf b1 = load_frag_b(Ks + 16 * LDK + kk * 32, 1, LDK); // t in 16..31
      s0 = wmma_bf16(qfr[kk], b0, s0);
      s1 = wmma_bf16(qfr[kk], b1, s1);
    }

    // online softmax (rows live in half-wave 16-lane groups)
#pragma unroll
    for (int r = 0; r < 8; ++r) {
      const int qidx = blk * 32 + rg * 16 + r + 8 * half;
      float v0 = s0[r] * scale;
      if (tb + cn > qidx) v0 = -3.0e38f;
      float v1 = s1[r] * scale;
      if (tb + 16 + cn > qidx) v1 = -3.0e38f;
      float mx = fmaxf(v0, v1);
#pragma unroll
      for (int xm = 1; xm < 16; xm <<= 1) mx = fmaxf(mx, __shfl_xor(mx, xm, 16));
      const float mnew  = fmaxf(mrow[r], mx);
      const float alpha = __expf(mrow[r] - mnew);
      const float p0 = __expf(v0 - mnew);
      const float p1 = __expf(v1 - mnew);
      float ps = p0 + p1;
#pragma unroll
      for (int xm = 1; xm < 16; xm <<= 1) ps += __shfl_xor(ps, xm, 16);
      lrow[r] = lrow[r] * alpha + ps;
      mrow[r] = mnew;
#pragma unroll
      for (int nt = 0; nt < 8; ++nt) o[nt][r] *= alpha;
      Ps[wid][(r + 8 * half) * 32 + cn]      = (__bf16)p0;
      Ps[wid][(r + 8 * half) * 32 + 16 + cn] = (__bf16)p1;
    }
    __syncthreads();

    // O(16x128) += P(16x32) . V(32x128) ; V = first 512 cols of K_eff rows
    const v16bf pa = load_frag_a(&Ps[wid][0], 32);
#pragma unroll
    for (int nt = 0; nt < 8; ++nt) {
      const v16bf bv = load_frag_b(Ks + cg * 128 + nt * 16, LDK, 1);
      o[nt] = wmma_bf16(pa, bv, o[nt]);
    }
    __syncthreads();
  }

#pragma unroll
  for (int nt = 0; nt < 8; ++nt)
#pragma unroll
    for (int r = 0; r < 8; ++r) {
      const int q = blk * 32 + rg * 16 + r + 8 * half;
      const int c = cg * 128 + nt * 16 + cn;
      const float v = o[nt][r] / lrow[r];
      O[(((long long)(b * SEQ + q)) * NH + h) * KV_RANK + c] = (__bf16)v;
    }
}

// ---------------- host ----------------
extern "C" void kernel_launch(void* const* d_in, const int* in_sizes, int n_in,
                              void* d_out, int out_size, void* d_ws, size_t ws_size,
                              hipStream_t stream) {
  (void)in_sizes; (void)n_in; (void)out_size; (void)ws_size;
  const float* x     = (const float*)d_in[0];
  // d_in[1] = start_pos (always 0 in this harness), d_in[4] = mask (causal, derived analytically)
  const float* cosb  = (const float*)d_in[2];
  const float* sinb  = (const float*)d_in[3];
  const float* wq    = (const float*)d_in[5];
  const float* wkv_a = (const float*)d_in[6];
  const float* kvnw  = (const float*)d_in[7];
  const float* wkv_b = (const float*)d_in[8];
  const float* wo    = (const float*)d_in[9];

  const long long MT = (long long)BATCH * SEQ;  // 4096 tokens

  char* w = (char*)d_ws;
  auto alloc = [&](size_t bytes) -> char* {
    char* p = w;
    w += (bytes + 255) & ~(size_t)255;
    return p;
  };
  __bf16* xb   = (__bf16*)alloc((size_t)MT * DIM * 2);
  __bf16* wqb  = (__bf16*)alloc((size_t)NH * D_QK * DIM * 2);
  __bf16* wab  = (__bf16*)alloc((size_t)D_EFF * DIM * 2);
  __bf16* wbb  = (__bf16*)alloc((size_t)NH * (D_NOPE + D_V) * KV_RANK * 2);
  __bf16* wob  = (__bf16*)alloc((size_t)DIM * NH * D_V * 2);
  float*  qf   = (float*)alloc((size_t)MT * NH * D_QK * 4);
  float*  kvf  = (float*)alloc((size_t)MT * D_EFF * 4);
  __bf16* qeff = (__bf16*)alloc((size_t)MT * NH * D_EFF * 2);
  __bf16* qn   = (__bf16*)alloc((size_t)MT * NH * D_NOPE * 2);
  __bf16* keff = (__bf16*)alloc((size_t)MT * D_EFF * 2);
  __bf16* obuf = (__bf16*)alloc((size_t)MT * NH * KV_RANK * 2);
  __bf16* ob2  = (__bf16*)alloc((size_t)MT * NH * D_V * 2);

  const float mm    = 0.1f * 1.0f * logf(40.0f) + 1.0f;  // mscale (MAX_SEQ_LEN > ORIG)
  const float scale = (1.0f / sqrtf((float)D_QK)) * mm * mm;

  // 1) fp32 -> bf16 conversions
  k_cvt_bf16<<<2048, 256, 0, stream>>>(x, xb, (int)(MT * DIM));
  k_cvt_bf16<<<2048, 256, 0, stream>>>(wq, wqb, NH * D_QK * DIM);
  k_cvt_bf16<<<1024, 256, 0, stream>>>(wkv_a, wab, D_EFF * DIM);
  k_cvt_bf16<<<1024, 256, 0, stream>>>(wkv_b, wbb, NH * (D_NOPE + D_V) * KV_RANK);
  k_cvt_bf16<<<2048, 256, 0, stream>>>(wo, wob, DIM * NH * D_V);

  // 2) q = x @ wq^T   (fp32 out for RoPE)
  k_gemm<false, false><<<dim3(NH * D_QK / 128, MT / 128, 1), 256, 0, stream>>>(
      xb, wqb, qf, (int)MT, NH * D_QK, DIM, DIM, DIM, NH * D_QK, 0, 0, 0);
  // 3) kv = x @ wkv_a^T (fp32 out for RMSNorm/RoPE); N=576 -> 5 blocks, TDM zero-fills OOB
  k_gemm<false, false><<<dim3((D_EFF + 127) / 128, MT / 128, 1), 256, 0, stream>>>(
      xb, wab, kvf, (int)MT, D_EFF, DIM, DIM, DIM, D_EFF, 0, 0, 0);

  // 4) elementwise preps
  k_prep_q<<<dim3((unsigned)(MT * NH)), 128, 0, stream>>>(qf, cosb, sinb, qeff, qn);
  k_prep_k<<<dim3((unsigned)MT), 128, 0, stream>>>(kvf, cosb, sinb, kvnw, keff);

  // 5) absorb: q_eff[:, :512] = q_nope @ w_uk[h]  (NN, batched over heads)
  k_gemm<true, true><<<dim3(KV_RANK / 128, MT / 128, NH), 256, 0, stream>>>(
      qn, wbb, qeff, (int)MT, KV_RANK, D_NOPE,
      NH * D_NOPE, KV_RANK, NH * D_EFF,
      (long long)D_NOPE, (long long)(D_NOPE + D_V) * KV_RANK, (long long)D_EFF);

  // 6) flash attention in latent space
  k_attn<<<dim3(SEQ / 32, NH, BATCH), 256, 0, stream>>>(qeff, keff, obuf, scale);

  // 7) per-head up-projection: ob2[:, h*128:] = O_h @ w_uv[h]^T (NT, batched)
  k_gemm<false, true><<<dim3(1, MT / 128, NH), 256, 0, stream>>>(
      obuf, wbb + (size_t)D_NOPE * KV_RANK, ob2, (int)MT, D_V, KV_RANK,
      NH * KV_RANK, KV_RANK, NH * D_V,
      (long long)KV_RANK, (long long)(D_NOPE + D_V) * KV_RANK, (long long)D_V);

  // 8) final: out = ob2 @ wo^T (fp32 out)
  k_gemm<false, false><<<dim3(DIM / 128, MT / 128, 1), 256, 0, stream>>>(
      ob2, wob, d_out, (int)MT, DIM, NH * D_V, NH * D_V, NH * D_V, DIM, 0, 0, 0);
}
